// Pointnet2MSG_67448166416421
// MI455X (gfx1250) — compile-verified, matrix operand order checked by reading, not run on hardware
//
#include <hip/hip_runtime.h>

// ---------------------------------------------------------------------------
// Types for CDNA5 WMMA
// ---------------------------------------------------------------------------
typedef __attribute__((ext_vector_type(16))) _Float16 v16h;
typedef __attribute__((ext_vector_type(8)))  _Float16 v8h;
typedef __attribute__((ext_vector_type(8)))  float    v8f;

// Load one f16 WMMA fragment (A or B side) for a 16x32 (MxK) tile.
// ISA layout (CDNA5 ISA, 16-bit A 16x32): lane%16 = row,
// lane/16 selects K-halves: half0 -> K {0..7, 16..23}, half1 -> K {8..15, 24..31}.
// Caller passes p = base + row*ld + (lane>>4)*8; we read K {0..7} and K {16..23}
// relative to that, which yields exactly the required packing for both halves.
static __device__ __forceinline__ v16h load_frag16(const _Float16* p) {
    v8h lo = *(const v8h*)(p);
    v8h hi = *(const v8h*)(p + 16);
    v16h r;
#pragma unroll
    for (int i = 0; i < 8; ++i) { r[i] = lo[i]; r[i + 8] = hi[i]; }
    return r;
}

// ---------------------------------------------------------------------------
// Generic fused GEMM:  O = relu((A . W^T) * s + b)
//   A:  R x lda   (f16, zero padded to lda = ceil32(K))
//   W:  Np x ldw  (f16, zero padded rows to ceil32(cout), cols to ldw == lda)
//   O:  R x ldo   (f16, ldo = ceil32(cout); cols >= cout are written as 0)
//   optional f32 secondary output:
//     o32mode 1: O32[m*ld32 + n]                (row major, valid n only)
//     o32mode 2: O32[(b*cout + n)*npts + m%npts] (transposed (B,C,N) final out)
// One wave computes a 64x16 output tile (4 WMMA accumulators share B fragment).
// The K loop is software pipelined: fragments for chunk k+32 are issued before
// the 4 WMMAs of chunk k, so global-load latency overlaps matrix execution
// instead of a full s_wait_loadcnt 0 stall in front of every WMMA.
// R must be a multiple of 64 (true for every layer in this net).
// ---------------------------------------------------------------------------
__global__ __launch_bounds__(256) void k_gemm_wmma(
    const _Float16* __restrict__ A, int lda,
    const _Float16* __restrict__ W, int ldw,
    const float* __restrict__ scale, const float* __restrict__ bias,
    _Float16* __restrict__ O, int ldo,
    float* __restrict__ O32, int o32mode, int ld32, int npts,
    int R, int cout)
{
    const int lane   = threadIdx.x & 31;
    const int wid    = (int)((blockIdx.x * blockDim.x + threadIdx.x) >> 5);
    const int tilesN = ldo >> 4;
    const int tilesM = R >> 6;
    if (wid >= tilesM * tilesN) return;

    const int tn = wid % tilesN;
    const int tm = wid / tilesN;
    const int n0 = tn << 4;
    const int m0 = tm << 6;
    const int lrow  = lane & 15;
    const int khalf = lane >> 4;

    v8f zero = {};
    v8f acc[4];
#pragma unroll
    for (int i = 0; i < 4; ++i) acc[i] = zero;

    const _Float16* Wp = W + (size_t)(n0 + lrow) * ldw + khalf * 8;
    const _Float16* Ap = A + (size_t)(m0 + lrow) * lda + khalf * 8;
    const size_t mstep = (size_t)16 * lda;

    // ---- software-pipelined K loop (register double buffering) ----
    v16h bf  = load_frag16(Wp);
    v16h af0 = load_frag16(Ap);
    v16h af1 = load_frag16(Ap + mstep);
    v16h af2 = load_frag16(Ap + 2 * mstep);
    v16h af3 = load_frag16(Ap + 3 * mstep);

    for (int k0 = 0; k0 < lda; k0 += 32) {
        v16h bn = bf, an0 = af0, an1 = af1, an2 = af2, an3 = af3;
        const int k1 = k0 + 32;
        if (k1 < lda) {
            bn  = load_frag16(Wp + k1);
            an0 = load_frag16(Ap + k1);
            an1 = load_frag16(Ap + mstep + k1);
            an2 = load_frag16(Ap + 2 * mstep + k1);
            an3 = load_frag16(Ap + 3 * mstep + k1);
        }
        acc[0] = __builtin_amdgcn_wmma_f32_16x16x32_f16(
            false, af0, false, bf, (short)0, acc[0], false, false);
        acc[1] = __builtin_amdgcn_wmma_f32_16x16x32_f16(
            false, af1, false, bf, (short)0, acc[1], false, false);
        acc[2] = __builtin_amdgcn_wmma_f32_16x16x32_f16(
            false, af2, false, bf, (short)0, acc[2], false, false);
        acc[3] = __builtin_amdgcn_wmma_f32_16x16x32_f16(
            false, af3, false, bf, (short)0, acc[3], false, false);
        bf = bn; af0 = an0; af1 = an1; af2 = an2; af3 = an3;
    }

    const int  nn    = n0 + lrow;
    const bool valid = nn < cout;
    const float sv = valid ? scale[nn] : 0.0f;
    const float bv = valid ? bias[nn]  : 0.0f;

#pragma unroll
    for (int mt = 0; mt < 4; ++mt) {
#pragma unroll
        for (int r = 0; r < 8; ++r) {
            // D layout: lanes 0..15 -> M = r, lanes 16..31 -> M = r + 8
            int m = m0 + mt * 16 + r + 8 * khalf;
            float y = valid ? fmaxf(acc[mt][r] * sv + bv, 0.0f) : 0.0f;
            O[(size_t)m * ldo + nn] = (_Float16)y;
            if (o32mode == 1) {
                if (valid) O32[(size_t)m * ld32 + nn] = y;
            } else if (o32mode == 2) {
                if (valid) {
                    int b = m / npts, n = m % npts;
                    O32[((size_t)b * cout + nn) * npts + n] = y;
                }
            }
        }
    }
}

// ---------------------------------------------------------------------------
// Split pointcloud (B,N,9) -> xyz (B,N,3) + feat (B,N,6)
// ---------------------------------------------------------------------------
__global__ void k_split(const float* __restrict__ pc, float* __restrict__ xyz,
                        float* __restrict__ feat, long total /*B*N*/)
{
    long t = blockIdx.x * (long)blockDim.x + threadIdx.x;
    if (t >= total) return;
    const float* p = pc + t * 9;
#pragma unroll
    for (int c = 0; c < 3; ++c) xyz[t * 3 + c] = p[c];
#pragma unroll
    for (int c = 0; c < 6; ++c) feat[t * 6 + c] = p[3 + c];
}

// ---------------------------------------------------------------------------
// Convert a (cout x cin) f32 weight to zero-padded f16 (Cp cols, total elems)
// ---------------------------------------------------------------------------
__global__ void k_convw(const float* __restrict__ W, int cout, int cin,
                        _Float16* __restrict__ Wh, int Cp, long total)
{
    long e = blockIdx.x * (long)blockDim.x + threadIdx.x;
    if (e >= total) return;
    int r = (int)(e / Cp), c = (int)(e % Cp);
    float v = (r < cout && c < cin) ? W[(size_t)r * cin + c] : 0.0f;
    Wh[e] = (_Float16)v;
}

// ---------------------------------------------------------------------------
// Farthest point sampling; one block per batch, dists in global scratch.
// Matches reference: fidx[0] = 0; argmax with lowest-index tie break.
// ---------------------------------------------------------------------------
__global__ __launch_bounds__(256) void k_fps(const float* __restrict__ xyz,
                                             float* __restrict__ dists,
                                             int N, int M, int* __restrict__ out)
{
    const int b = blockIdx.x;
    const int t = threadIdx.x;
    const float* X = xyz + (size_t)b * N * 3;
    float* D = dists + (size_t)b * N;
    __shared__ float rv[256];
    __shared__ int   ri[256];
    __shared__ int   sfar;
    for (int n = t; n < N; n += 256) D[n] = 1.0e10f;
    if (t == 0) sfar = 0;
    __syncthreads();
    for (int p = 0; p < M; ++p) {
        int far = sfar;
        if (t == 0) out[b * M + p] = far;
        float cx = X[far * 3 + 0], cy = X[far * 3 + 1], cz = X[far * 3 + 2];
        float bm = -1.0f; int ba = 0x7fffffff;
        for (int n = t; n < N; n += 256) {
            float dx = X[n * 3 + 0] - cx;
            float dy = X[n * 3 + 1] - cy;
            float dz = X[n * 3 + 2] - cz;
            float d  = dx * dx + dy * dy + dz * dz;
            float nd = fminf(D[n], d);
            D[n] = nd;
            if (nd > bm) { bm = nd; ba = n; }
        }
        rv[t] = bm; ri[t] = ba;
        __syncthreads();
        for (int s = 128; s > 0; s >>= 1) {
            if (t < s) {
                float ov = rv[t + s]; int oi = ri[t + s];
                if (ov > rv[t] || (ov == rv[t] && oi < ri[t])) { rv[t] = ov; ri[t] = oi; }
            }
            __syncthreads();
        }
        if (t == 0) sfar = ri[0];
        __syncthreads();
    }
}

// ---------------------------------------------------------------------------
// Gather sampled centroids: out[b,m,:] = xyz[b, fidx[b,m], :]
// ---------------------------------------------------------------------------
__global__ void k_gather_xyz(const float* __restrict__ xyz, const int* __restrict__ fidx,
                             int N, int M, float* __restrict__ out, long total /*B*M*/)
{
    long t = blockIdx.x * (long)blockDim.x + threadIdx.x;
    if (t >= total) return;
    int b = (int)(t / M);
    int n = fidx[t];
    const float* s = xyz + ((size_t)b * N + n) * 3;
#pragma unroll
    for (int c = 0; c < 3; ++c) out[t * 3 + c] = s[c];
}

// ---------------------------------------------------------------------------
// Ball query: first `ns` indices (ascending) with d2 <= r2; pad with first (or 0).
// ---------------------------------------------------------------------------
__global__ void k_ball(const float* __restrict__ xyz, const float* __restrict__ nxyz,
                       int N, int M, float r2, int ns, int* __restrict__ idx, long total)
{
    long t = blockIdx.x * (long)blockDim.x + threadIdx.x;
    if (t >= total) return;
    int b = (int)(t / M);
    const float* q = nxyz + (size_t)t * 3;
    float qx = q[0], qy = q[1], qz = q[2];
    const float* X = xyz + (size_t)b * N * 3;
    int* o = idx + (size_t)t * ns;
    int cnt = 0;
    for (int n = 0; n < N && cnt < ns; ++n) {
        float dx = X[n * 3 + 0] - qx;
        float dy = X[n * 3 + 1] - qy;
        float dz = X[n * 3 + 2] - qz;
        if (dx * dx + dy * dy + dz * dz <= r2) o[cnt++] = n;
    }
    int first = (cnt > 0) ? o[0] : 0;
    for (; cnt < ns; ++cnt) o[cnt] = first;
}

// ---------------------------------------------------------------------------
// Build grouped SA input (f16, zero padded to Kp):
//   row = ((b*M+m)*ns+s), cols [0:3) = xyz[idx]-new_xyz, [3:3+Cf) = feat[idx]
// ---------------------------------------------------------------------------
__global__ void k_group(const int* __restrict__ idx, const float* __restrict__ xyz,
                        const float* __restrict__ nxyz, const float* __restrict__ feat,
                        int ldf, int N, int M, int ns, int Cf, int Kp,
                        _Float16* __restrict__ A, long total)
{
    long e = blockIdx.x * (long)blockDim.x + threadIdx.x;
    if (e >= total) return;
    long row = e / Kp;
    int  c   = (int)(e % Kp);
    long bm  = row / ns;
    int  m   = (int)(bm % M);
    int  b   = (int)(bm / M);
    float v  = 0.0f;
    int n = idx[row];
    if (c < 3) {
        v = xyz[((size_t)b * N + n) * 3 + c] - nxyz[((size_t)b * M + m) * 3 + c];
    } else if (c < 3 + Cf) {
        v = feat[((size_t)b * N + n) * ldf + (c - 3)];
    }
    A[e] = (_Float16)v;
}

// ---------------------------------------------------------------------------
// Max over the ns samples of the last SA layer output -> f32 level features.
// ---------------------------------------------------------------------------
__global__ void k_maxpool(const _Float16* __restrict__ A, int ldo, int ns, int cout,
                          float* __restrict__ feat, int ldf, int coff, long total)
{
    long t = blockIdx.x * (long)blockDim.x + threadIdx.x;
    if (t >= total) return;
    int  c  = (int)(t % cout);
    long bm = t / cout;
    const _Float16* p = A + (size_t)bm * ns * ldo + c;
    float mx = -3.4e38f;
    for (int s = 0; s < ns; ++s) mx = fmaxf(mx, (float)p[(size_t)s * ldo]);
    feat[(size_t)bm * ldf + coff + c] = mx;
}

// ---------------------------------------------------------------------------
// 3-NN (by squared distance, lowest-index tie break) + inverse-d2 weights.
// ---------------------------------------------------------------------------
__global__ void k_knn3(const float* __restrict__ unk, const float* __restrict__ kn,
                       int Nu, int Nk, int* __restrict__ idx3, float* __restrict__ w3,
                       long total /*B*Nu*/)
{
    long t = blockIdx.x * (long)blockDim.x + threadIdx.x;
    if (t >= total) return;
    int b = (int)(t / Nu);
    const float* P = unk + (size_t)t * 3;
    float px = P[0], py = P[1], pz = P[2];
    const float* K = kn + (size_t)b * Nk * 3;
    float d0 = 3.4e38f, d1 = 3.4e38f, d2 = 3.4e38f;
    int   i0 = 0, i1 = 0, i2 = 0;
    for (int n = 0; n < Nk; ++n) {
        float dx = K[n * 3 + 0] - px;
        float dy = K[n * 3 + 1] - py;
        float dz = K[n * 3 + 2] - pz;
        float d  = dx * dx + dy * dy + dz * dz;
        if (d < d0)      { d2 = d1; i2 = i1; d1 = d0; i1 = i0; d0 = d; i0 = n; }
        else if (d < d1) { d2 = d1; i2 = i1; d1 = d;  i1 = n; }
        else if (d < d2) { d2 = d;  i2 = n; }
    }
    float r0 = 1.0f / (fmaxf(d0, 0.0f) + 1e-8f);
    float r1 = 1.0f / (fmaxf(d1, 0.0f) + 1e-8f);
    float r2 = 1.0f / (fmaxf(d2, 0.0f) + 1e-8f);
    float inv = 1.0f / (r0 + r1 + r2);
    idx3[t * 3 + 0] = i0; idx3[t * 3 + 1] = i1; idx3[t * 3 + 2] = i2;
    w3[t * 3 + 0] = r0 * inv; w3[t * 3 + 1] = r1 * inv; w3[t * 3 + 2] = r2 * inv;
}

// ---------------------------------------------------------------------------
// Build FP input (f16, zero padded to Kp):
//   cols [0:Ck) = 3-NN interpolated known feats, [Ck:Ck+Cu) = skip feats.
// ---------------------------------------------------------------------------
__global__ void k_interp(const float* __restrict__ featK, int ldfk, int Ck, int Nk,
                         const float* __restrict__ featU, int ldfu, int Cu,
                         const int* __restrict__ idx3, const float* __restrict__ w3,
                         int Nu, int Kp, _Float16* __restrict__ A, long total)
{
    long e = blockIdx.x * (long)blockDim.x + threadIdx.x;
    if (e >= total) return;
    long row = e / Kp;
    int  c   = (int)(e % Kp);
    int  b   = (int)(row / Nu);
    float v  = 0.0f;
    if (c < Ck) {
        const float* base = featK + (size_t)b * Nk * ldfk;
        const int*   I = idx3 + row * 3;
        const float* W = w3   + row * 3;
        v = base[(size_t)I[0] * ldfk + c] * W[0] +
            base[(size_t)I[1] * ldfk + c] * W[1] +
            base[(size_t)I[2] * ldfk + c] * W[2];
    } else if (c < Ck + Cu) {
        v = featU[row * (size_t)ldfu + (c - Ck)];
    }
    A[e] = (_Float16)v;
}

// ---------------------------------------------------------------------------
// Host orchestration
// ---------------------------------------------------------------------------
static inline unsigned gs(long n) { return (unsigned)((n + 255) / 256); }
static inline int c32(int x) { return (x + 31) & ~31; }

extern "C" void kernel_launch(void* const* d_in, const int* in_sizes, int n_in,
                              void* d_out, int out_size, void* d_ws, size_t ws_size,
                              hipStream_t stream)
{
    (void)in_sizes; (void)n_in; (void)out_size; (void)ws_size;
    const int B = 8;
    static const int Ns[5]      = {16384, 512, 256, 128, 64};
    static const int NPOINT[4]  = {512, 256, 128, 64};
    static const int NSTAB[2]   = {16, 32};
    static const float R2TAB[4][2] = {{1.0e-4f, 4.0e-4f}, {4.0e-4f, 1.6e-3f},
                                      {1.6e-3f, 6.4e-3f}, {6.4e-3f, 2.56e-2f}};
    static const int SA_CIN[8][3]  = {{9,16,16},{9,32,32},{99,64,64},{99,64,96},
                                      {259,128,196},{259,128,196},{515,256,256},{515,256,384}};
    static const int SA_COUT[8][3] = {{16,16,32},{32,32,64},{64,64,128},{64,96,128},
                                      {128,196,256},{128,196,256},{256,256,512},{256,384,512}};
    static const int FP_CIN[4][2]  = {{134,64},{352,128},{768,256},{1536,512}};
    static const int FP_COUT[4][2] = {{64,64},{128,128},{256,256},{512,512}};
    static const int LDF[5] = {6, 128, 256, 512, 1024};

    const float* pc = (const float*)d_in[0];

    // ---- input pointer tables (dict insertion order: pointcloud, sa, fp) ----
    const float *saW[8][3], *saS[8][3], *saB[8][3];
    const float *fpW[4][2], *fpS[4][2], *fpB[4][2];
    {
        int p = 1;
        for (int br = 0; br < 8; ++br)
            for (int l = 0; l < 3; ++l) {
                saW[br][l] = (const float*)d_in[p++];
                saS[br][l] = (const float*)d_in[p++];
                saB[br][l] = (const float*)d_in[p++];
            }
        for (int i = 0; i < 4; ++i)
            for (int l = 0; l < 2; ++l) {
                fpW[i][l] = (const float*)d_in[p++];
                fpS[i][l] = (const float*)d_in[p++];
                fpB[i][l] = (const float*)d_in[p++];
            }
    }

    // ---- workspace carving ----
    char* w = (char*)d_ws;
    auto alloc = [&](size_t bytes) -> void* {
        void* r = (void*)w;
        w += (bytes + 255) & ~(size_t)255;
        return r;
    };
    float* xyzL[5];
    float* featL[5];
    for (int k = 0; k < 5; ++k) xyzL[k]  = (float*)alloc((size_t)B * Ns[k] * 3 * 4);
    for (int k = 0; k < 5; ++k) featL[k] = (float*)alloc((size_t)B * Ns[k] * LDF[k] * 4);
    float* dists = (float*)alloc((size_t)B * 16384 * 4);
    int*   fidx  = (int*)alloc((size_t)B * 512 * 4);
    int*   bidx  = (int*)alloc((size_t)B * 512 * 32 * 4);
    int*   idx3  = (int*)alloc((size_t)B * 16384 * 3 * 4);
    float* w3    = (float*)alloc((size_t)B * 16384 * 3 * 4);

    _Float16* saWh[8][3];
    _Float16* fpWh[4][2];
    for (int br = 0; br < 8; ++br)
        for (int l = 0; l < 3; ++l) {
            int Rp = c32(SA_COUT[br][l]), Cp = c32(SA_CIN[br][l]);
            size_t tot = (size_t)Rp * Cp;
            saWh[br][l] = (_Float16*)alloc(tot * 2);
            k_convw<<<gs((long)tot), 256, 0, stream>>>(saW[br][l], SA_COUT[br][l],
                                                       SA_CIN[br][l], saWh[br][l], Cp, (long)tot);
        }
    for (int i = 0; i < 4; ++i)
        for (int l = 0; l < 2; ++l) {
            int Rp = c32(FP_COUT[i][l]), Cp = c32(FP_CIN[i][l]);
            size_t tot = (size_t)Rp * Cp;
            fpWh[i][l] = (_Float16*)alloc(tot * 2);
            k_convw<<<gs((long)tot), 256, 0, stream>>>(fpW[i][l], FP_COUT[i][l],
                                                       FP_CIN[i][l], fpWh[i][l], Cp, (long)tot);
        }

    _Float16* actA = (_Float16*)alloc((size_t)131072 * 160 * 2); // 40 MB
    _Float16* actB = (_Float16*)alloc((size_t)131072 * 64  * 2); // 16 MB

    // ---- split pointcloud ----
    k_split<<<gs((long)B * 16384), 256, 0, stream>>>(pc, xyzL[0], featL[0], (long)B * 16384);

    int curCh[5] = {6, 96, 256, 512, 1024}; // feature widths (valid after each SA level)

    // ---- set-abstraction levels ----
    for (int i = 0; i < 4; ++i) {
        int N = Ns[i], M = NPOINT[i];
        k_fps<<<B, 256, 0, stream>>>(xyzL[i], dists, N, M, fidx);
        k_gather_xyz<<<gs((long)B * M), 256, 0, stream>>>(xyzL[i], fidx, N, M, xyzL[i + 1],
                                                          (long)B * M);
        int coff = 0;
        for (int j = 0; j < 2; ++j) {
            int br = i * 2 + j;
            int ns = NSTAB[j];
            long rows = (long)B * M * ns;
            k_ball<<<gs((long)B * M), 256, 0, stream>>>(xyzL[i], xyzL[i + 1], N, M,
                                                        R2TAB[i][j], ns, bidx, (long)B * M);
            int Cf = curCh[i];
            int Kp = c32(3 + Cf);
            k_group<<<gs(rows * Kp), 256, 0, stream>>>(bidx, xyzL[i], xyzL[i + 1], featL[i],
                                                       LDF[i], N, M, ns, Cf, Kp, actA,
                                                       rows * Kp);
            const _Float16* cur = actA;
            int curK = Kp;
            int lastOut = 0;
            for (int l = 0; l < 3; ++l) {
                int cout = SA_COUT[br][l];
                int ldo  = c32(cout);
                _Float16* dst = (l % 2 == 0) ? actB : actA;
                long nwaves = (rows >> 6) * (ldo >> 4);
                k_gemm_wmma<<<(unsigned)((nwaves + 7) / 8), 256, 0, stream>>>(
                    cur, curK, saWh[br][l], curK, saS[br][l], saB[br][l],
                    dst, ldo, nullptr, 0, 0, 0, (int)rows, cout);
                cur = dst; curK = ldo; lastOut = cout;
            }
            k_maxpool<<<gs((long)B * M * lastOut), 256, 0, stream>>>(
                cur, curK, ns, lastOut, featL[i + 1], LDF[i + 1], coff, (long)B * M * lastOut);
            coff += lastOut;
        }
    }

    // ---- feature propagation (i = -1 .. -4  <=>  t = 0 .. 3, fp index 3-t) ----
    float* dout = (float*)d_out;
    for (int t = 0; t < 4; ++t) {
        int ui = 3 - t, ki = 4 - t;
        int Nu = Ns[ui], Nk = Ns[ki];
        int fp = 3 - t;
        k_knn3<<<gs((long)B * Nu), 256, 0, stream>>>(xyzL[ui], xyzL[ki], Nu, Nk, idx3, w3,
                                                     (long)B * Nu);
        int Ck = curCh[ki], Cu = curCh[ui];
        int Kp = c32(Ck + Cu);
        long rows = (long)B * Nu;
        k_interp<<<gs(rows * Kp), 256, 0, stream>>>(featL[ki], LDF[ki], Ck, Nk,
                                                    featL[ui], LDF[ui], Cu,
                                                    idx3, w3, Nu, Kp, actA, rows * Kp);
        // layer 0
        int c0 = FP_COUT[fp][0], ldo0 = c32(c0);
        {
            long nwaves = (rows >> 6) * (ldo0 >> 4);
            k_gemm_wmma<<<(unsigned)((nwaves + 7) / 8), 256, 0, stream>>>(
                actA, Kp, fpWh[fp][0], Kp, fpS[fp][0], fpB[fp][0],
                actB, ldo0, nullptr, 0, 0, 0, (int)rows, c0);
        }
        // layer 1
        int c1 = FP_COUT[fp][1], ldo1 = c32(c1);
        {
            long nwaves = (rows >> 6) * (ldo1 >> 4);
            if (t < 3) {
                k_gemm_wmma<<<(unsigned)((nwaves + 7) / 8), 256, 0, stream>>>(
                    actB, ldo0, fpWh[fp][1], ldo0, fpS[fp][1], fpB[fp][1],
                    actA, ldo1, featL[ui], 1, LDF[ui], 0, (int)rows, c1);
                curCh[ui] = c1;
            } else {
                // final layer: write d_out as (B, C=64, N=16384)
                k_gemm_wmma<<<(unsigned)((nwaves + 7) / 8), 256, 0, stream>>>(
                    actB, ldo0, fpWh[fp][1], ldo0, fpS[fp][1], fpB[fp][1],
                    actA, ldo1, dout, 2, 0, 16384, (int)rows, c1);
            }
        }
    }
}
